// EncoderModule_41068477284460
// MI455X (gfx1250) — matrix operand order repaired and checked by reading, not compiled, not run on hardware
//
#include <hip/hip_runtime.h>

// ---------------------------------------------------------------------------
// Types for CDNA5 WMMA
// ---------------------------------------------------------------------------
typedef __attribute__((ext_vector_type(16))) __bf16 v16bf;
typedef __attribute__((ext_vector_type(8)))  float  v8f;

union Frag { unsigned int u[8]; v16bf v; };

__device__ __forceinline__ unsigned short f2bf(float f) {
  unsigned int u = __float_as_uint(f);
  u += 0x7fffu + ((u >> 16) & 1u);     // round-to-nearest-even
  return (unsigned short)(u >> 16);
}

// Async memory->LDS 16B copy (CDNA5): per-lane LDS dest addr + global addr.
// Tracked by ASYNCcnt; completed via s_wait_asynccnt.
__device__ __forceinline__ void async_copy_b128(unsigned lds_addr,
                                                const void* gaddr) {
  unsigned long long ga = (unsigned long long)gaddr;
  asm volatile("global_load_async_to_lds_b128 %0, %1, off"
               :: "v"(lds_addr), "v"(ga) : "memory");
}
// Wait until <=4 of this wave's async ops remain (ASYNCcnt retires in order;
// 4 ops per tile => the older of two in-flight tiles has fully landed).
__device__ __forceinline__ void wait_async_le4() {
  asm volatile("s_wait_asynccnt 0x4" ::: "memory");
}
__device__ __forceinline__ void wait_async0() {
  asm volatile("s_wait_asynccnt 0x0" ::: "memory");
}

// ---------------------------------------------------------------------------
// GEMM: C[M,N] = A[M,K] (bf16, row-major) @ Bt[N,K]^T (bf16, row-major [N][K])
//       + bias[N] (+ resid[M,N] f32) ; optional exact-erf GELU; writes f32
//       and/or bf16.  Batched over blockIdx.z via strides.
// Block: 256 threads (8 wave32), tile 128x128, BK=32.
// LDS: 3 rotating buffers filled by GLOBAL_LOAD_ASYNC_TO_LDS_B128 with two
// tiles in flight; branch-free steady-state loop (drain peeled), exactly ONE
// barrier per K-tile.  Wave grid 4(M) x 2(N): 2x4 WMMA accumulators per wave.
// ---------------------------------------------------------------------------
__global__ __launch_bounds__(256) void gemm_bf16_wmma(
    const unsigned short* __restrict__ A,
    const unsigned short* __restrict__ Bt,
    const float* __restrict__ bias,
    const float* __restrict__ resid,
    float* __restrict__ Cf,
    unsigned short* __restrict__ Cb,
    int M, int N, int K, int act,
    long long btStride, long long biasStride, long long cStride)
{
  // 40-element row pitch: 80B = 20 banks; 20*i mod 64 distinct for 16 lanes
  // -> conflict-free fragment reads; rows stay 16B-aligned for b128 copies.
  __shared__ unsigned short As[3][128 * 40];
  __shared__ unsigned short Bs[3][128 * 40];

  const int z = blockIdx.z;
  Bt += (size_t)z * btStride;
  if (bias) bias += (size_t)z * biasStride;
  if (Cf)   Cf   += (size_t)z * cStride;
  if (Cb)   Cb   += (size_t)z * cStride;

  const int m0 = blockIdx.y * 128;
  const int n0 = blockIdx.x * 128;
  const int t    = threadIdx.x;
  const int lane = t & 31;
  const int wv   = t >> 5;
  const int wm   = wv & 3;        // wave row 0..3  (32 rows each)
  const int wn   = wv >> 2;       // wave col 0..1  (64 cols each)
  const int h    = lane >> 4;     // half-wave selector
  const int mr   = lane & 15;

  // Tile fill assignment: 512 16B chunks per operand; thread t owns chunks
  // t and t+256 (rows row0 and row0+64, same 16B segment).
  const int row0 = t >> 2;
  const int seg0 = t & 3;
  const unsigned la_base = (unsigned)(unsigned long long)
                               &As[0][row0 * 40 + seg0 * 8];
  const unsigned lb_base = (unsigned)(unsigned long long)
                               &Bs[0][row0 * 40 + seg0 * 8];
  constexpr unsigned BUFB   = 128u * 40u * 2u;   // bytes per buffer
  constexpr unsigned HALF64 = 64u * 40u * 2u;    // 64 rows of pitch

  const unsigned short* gA0 = A  + (size_t)(m0 + row0) * K + seg0 * 8;
  const unsigned short* gB0 = Bt + (size_t)(n0 + row0) * K + seg0 * 8;

  auto issue_tile = [&](int buf, int k0) {
    const unsigned short* a0 = gA0 + k0;
    const unsigned short* b0 = gB0 + k0;
    const unsigned la = la_base + (unsigned)buf * BUFB;
    const unsigned lb = lb_base + (unsigned)buf * BUFB;
    async_copy_b128(la,          a0);
    async_copy_b128(la + HALF64, a0 + (size_t)64 * K);
    async_copy_b128(lb,          b0);
    async_copy_b128(lb + HALF64, b0 + (size_t)64 * K);
  };

  const v8f zero = {0.f,0.f,0.f,0.f,0.f,0.f,0.f,0.f};
  v8f acc[2][4];
#pragma unroll
  for (int i = 0; i < 2; ++i)
#pragma unroll
    for (int j = 0; j < 4; ++j) acc[i][j] = zero;

  auto compute_tile = [&](int buf) {
    Frag a[2], b[4];
#pragma unroll
    for (int mi = 0; mi < 2; ++mi) {
      const int ar = wm * 32 + mi * 16 + mr;
#pragma unroll
      for (int i = 0; i < 8; ++i) {
        // A 16x32 bf16 layout: i<4 -> K=2i+8h ; i>=4 -> K=16+2(i-4)+8h
        const int k = ((i >> 2) << 4) + (h << 3) + ((i & 3) << 1);
        a[mi].u[i] = *(const unsigned int*)&As[buf][ar * 40 + k];
      }
    }
#pragma unroll
    for (int ni = 0; ni < 4; ++ni) {
      const int br = wn * 64 + ni * 16 + mr;   // n-index (row of Bt tile)
#pragma unroll
      for (int i = 0; i < 8; ++i) {
        // B 32x16 bf16 layout: i<4 -> K=2i+8h ; i>=4 -> K=16+2(i-4)+8h
        const int k = ((i >> 2) << 4) + (h << 3) + ((i & 3) << 1);
        b[ni].u[i] = *(const unsigned int*)&Bs[buf][br * 40 + k];
      }
    }
#pragma unroll
    for (int mi = 0; mi < 2; ++mi)
#pragma unroll
      for (int ni = 0; ni < 4; ++ni)
        acc[mi][ni] = __builtin_amdgcn_wmma_f32_16x16x32_bf16(
            false, a[mi].v, false, b[ni].v, (short)0, acc[mi][ni],
            false, false);
  };

  const int nIter = K >> 5;       // all GEMMs here have K >= 1024 -> nIter >= 32
  issue_tile(0, 0);
  issue_tile(1, 32);

  // Steady state: branch-free; tile `it` computed while tile `it+1` flies and
  // tile `it+2` is being issued.  Buffer (it+2)%3 == (it-1)%3 was last read in
  // iteration it-1, which precedes every wave's barrier signal of iteration
  // it, so refilling it after the barrier is safe.
  int buf = 0;                    // buffer holding the tile we compute next
  int nbuf = 2;                   // buffer the next issue targets
  for (int it = 0; it < nIter - 2; ++it) {
    wait_async_le4();
    __syncthreads();
    issue_tile(nbuf, (it + 2) << 5);
    compute_tile(buf);
    buf  = (buf  == 2) ? 0 : buf + 1;
    nbuf = (nbuf == 2) ? 0 : nbuf + 1;
  }
  // Drain: tile nIter-2 (one copy still in flight), then tile nIter-1.
  wait_async_le4();
  __syncthreads();
  compute_tile(buf);
  buf = (buf == 2) ? 0 : buf + 1;
  wait_async0();
  __syncthreads();
  compute_tile(buf);

  // Epilogue: C layout is M = r + 8h, N = lane&15 per 16x16 tile.
  const int m_base = m0 + wm * 32;
  const int n_base = n0 + wn * 64;
#pragma unroll
  for (int mi = 0; mi < 2; ++mi)
#pragma unroll
    for (int ni = 0; ni < 4; ++ni)
#pragma unroll
      for (int r = 0; r < 8; ++r) {
        const int gm = m_base + mi * 16 + (h << 3) + r;
        const int gn = n_base + ni * 16 + (lane & 15);
        float v = acc[mi][ni][r];
        if (bias)  v += bias[gn];
        if (resid) v += resid[(size_t)gm * N + gn];
        if (act)   v = 0.5f * v * (1.0f + erff(v * 0.70710678118654752f));
        const size_t o = (size_t)gm * N + gn;
        if (Cf) Cf[o] = v;
        if (Cb) Cb[o] = f2bf(v);
      }
}

// ---------------------------------------------------------------------------
// Row softmax over 4096-wide rows -> bf16 probabilities.
// ---------------------------------------------------------------------------
__global__ __launch_bounds__(256) void softmax_rows_4096(
    const float* __restrict__ S, unsigned short* __restrict__ P)
{
  const int row = blockIdx.x;
  const int t = threadIdx.x, lane = t & 31, wv = t >> 5;
  const float* sr = S + (size_t)row * 4096;
  unsigned short* pr = P + (size_t)row * 4096;

  float e[16];
  float mx = -3.4e38f;
#pragma unroll
  for (int i = 0; i < 16; ++i) { e[i] = sr[t + i * 256]; mx = fmaxf(mx, e[i]); }
#pragma unroll
  for (int o = 16; o; o >>= 1) mx = fmaxf(mx, __shfl_xor(mx, o, 32));
  __shared__ float rm[8], rs[8];
  if (!lane) rm[wv] = mx;
  __syncthreads();
  float bm = rm[0];
#pragma unroll
  for (int j = 1; j < 8; ++j) bm = fmaxf(bm, rm[j]);

  float s = 0.f;
#pragma unroll
  for (int i = 0; i < 16; ++i) { e[i] = __expf(e[i] - bm); s += e[i]; }
#pragma unroll
  for (int o = 16; o; o >>= 1) s += __shfl_xor(s, o, 32);
  if (!lane) rs[wv] = s;
  __syncthreads();
  float ts = 0.f;
#pragma unroll
  for (int j = 0; j < 8; ++j) ts += rs[j];
  const float inv = 1.0f / ts;
#pragma unroll
  for (int i = 0; i < 16; ++i) pr[t + i * 256] = f2bf(e[i] * inv);
}

// ---------------------------------------------------------------------------
// LayerNorm over 2048-wide rows of (Xa [+ Xb]) -> optional f32 and bf16 out.
// ---------------------------------------------------------------------------
__global__ __launch_bounds__(256) void layernorm_rows_2048(
    const float* __restrict__ Xa, const float* __restrict__ Xb,
    const float* __restrict__ g,  const float* __restrict__ bb,
    float* __restrict__ outF, unsigned short* __restrict__ outB)
{
  const int row = blockIdx.x;
  const int t = threadIdx.x, lane = t & 31, wv = t >> 5;
  const size_t base = (size_t)row * 2048;

  float x[8];
  float s = 0.f, s2 = 0.f;
#pragma unroll
  for (int i = 0; i < 8; ++i) {
    const int c = t + i * 256;
    float v = Xa[base + c];
    if (Xb) v += Xb[base + c];
    x[i] = v; s += v; s2 += v * v;
  }
#pragma unroll
  for (int o = 16; o; o >>= 1) { s += __shfl_xor(s, o, 32); s2 += __shfl_xor(s2, o, 32); }
  __shared__ float rs[8], rq[8];
  if (!lane) { rs[wv] = s; rq[wv] = s2; }
  __syncthreads();
  float ts = 0.f, tq = 0.f;
#pragma unroll
  for (int j = 0; j < 8; ++j) { ts += rs[j]; tq += rq[j]; }
  const float mu  = ts * (1.0f / 2048.0f);
  const float var = tq * (1.0f / 2048.0f) - mu * mu;
  const float inv = rsqrtf(var + 1e-5f);
#pragma unroll
  for (int i = 0; i < 8; ++i) {
    const int c = t + i * 256;
    const float y = (x[i] - mu) * inv * g[c] + bb[c];
    if (outF) outF[base + c] = y;
    if (outB) outB[base + c] = f2bf(y);
  }
}

// ---------------------------------------------------------------------------
// Pack z_stack [8,4096,256] -> x [4096,2048] (f32 + bf16).
// ---------------------------------------------------------------------------
__global__ __launch_bounds__(256) void pack_x_kernel(
    const float* __restrict__ z, float* __restrict__ X32,
    unsigned short* __restrict__ Xb)
{
  const size_t idx = (size_t)blockIdx.x * 256 + threadIdx.x;  // 8M total
  const int b = (int)(idx >> 11);
  const int c = (int)(idx & 2047);
  const int m = c >> 8, zz = c & 255;
  const float v = z[((size_t)m * 4096 + b) * 256 + zz];
  X32[idx] = v;
  Xb[idx]  = f2bf(v);
}

// ---------------------------------------------------------------------------
// Transpose src f32 [R][C] -> dst bf16 [C][R], batched via blockIdx.z.
// ---------------------------------------------------------------------------
__global__ __launch_bounds__(256) void transpose_f32_to_bf16(
    const float* __restrict__ src, unsigned short* __restrict__ dst,
    int R, int C, long long sStride, long long dStride)
{
  src += (size_t)blockIdx.z * sStride;
  dst += (size_t)blockIdx.z * dStride;
  __shared__ float tile[32][33];
  const int c0 = blockIdx.x * 32, r0 = blockIdx.y * 32;
  const int tx = threadIdx.x, ty = threadIdx.y;  // 32 x 8
#pragma unroll
  for (int j = 0; j < 4; ++j)
    tile[ty + j * 8][tx] = src[(size_t)(r0 + ty + j * 8) * C + c0 + tx];
  __syncthreads();
#pragma unroll
  for (int j = 0; j < 4; ++j)
    dst[(size_t)(c0 + ty + j * 8) * R + r0 + tx] = f2bf(tile[tx][ty + j * 8]);
}

// ---------------------------------------------------------------------------
// Transpose src bf16 [R][C] -> dst bf16 [C][R].
// ---------------------------------------------------------------------------
__global__ __launch_bounds__(256) void transpose_bf16(
    const unsigned short* __restrict__ src, unsigned short* __restrict__ dst,
    int R, int C)
{
  __shared__ unsigned short tile[32][33];
  const int c0 = blockIdx.x * 32, r0 = blockIdx.y * 32;
  const int tx = threadIdx.x, ty = threadIdx.y;  // 32 x 8
#pragma unroll
  for (int j = 0; j < 4; ++j)
    tile[ty + j * 8][tx] = src[(size_t)(r0 + ty + j * 8) * C + c0 + tx];
  __syncthreads();
#pragma unroll
  for (int j = 0; j < 4; ++j)
    dst[(size_t)(c0 + ty + j * 8) * R + r0 + tx] = tile[tx][ty + j * 8];
}

// ---------------------------------------------------------------------------
// Host-side orchestration.
// ---------------------------------------------------------------------------
extern "C" void kernel_launch(void* const* d_in, const int* in_sizes, int n_in,
                              void* d_out, int out_size, void* d_ws, size_t ws_size,
                              hipStream_t stream) {
  (void)in_sizes; (void)n_in; (void)out_size; (void)ws_size;

  const float* z_stack = (const float*)d_in[0];
  const float* Wq  = (const float*)d_in[1];
  const float* bq  = (const float*)d_in[2];
  const float* Wk  = (const float*)d_in[3];
  const float* bk  = (const float*)d_in[4];
  const float* Wv  = (const float*)d_in[5];
  const float* bv  = (const float*)d_in[6];
  const float* W1  = (const float*)d_in[7];
  const float* b1  = (const float*)d_in[8];
  const float* W2  = (const float*)d_in[9];
  const float* b2  = (const float*)d_in[10];
  const float* ln1_g = (const float*)d_in[11];
  const float* ln1_b = (const float*)d_in[12];
  const float* ln2_g = (const float*)d_in[13];
  const float* ln2_b = (const float*)d_in[14];
  const float* Wys = (const float*)d_in[15];
  const float* bys = (const float*)d_in[16];
  const float* Ws  = (const float*)d_in[17];
  const float* bs  = (const float*)d_in[18];
  float* out = (float*)d_out;

  constexpr size_t MB = 1ull << 20;
  char* ws = (char*)d_ws;
  // Persistent (per-launch) bf16 transposed weights: [0, 40MB)
  unsigned short* WqT  = (unsigned short*)(ws + 0 * MB);    // [2048][2048] 8MB
  unsigned short* WkT  = (unsigned short*)(ws + 8 * MB);    // 8MB
  unsigned short* WvT  = (unsigned short*)(ws + 16 * MB);   // 8MB
  unsigned short* W1T  = (unsigned short*)(ws + 24 * MB);   // [1024][2048] 4MB
  unsigned short* W2T  = (unsigned short*)(ws + 28 * MB);   // [2048][1024] 4MB
  unsigned short* WysT = (unsigned short*)(ws + 32 * MB);   // [1024][2048] 4MB
  unsigned short* WsT  = (unsigned short*)(ws + 36 * MB);   // [8][256][1024] 4MB
  // Activations (with lifetime-based reuse); peak = 248MB
  float*          X32  = (float*)         (ws + 40 * MB);   // [4096][2048] 32MB
  unsigned short* Xb   = (unsigned short*)(ws + 72 * MB);   // 16MB
  unsigned short* Qb   = (unsigned short*)(ws + 88 * MB);   // 16MB
  unsigned short* Kb   = (unsigned short*)(ws + 104 * MB);  // 16MB
  unsigned short* Vb   = (unsigned short*)(ws + 120 * MB);  // 16MB
  unsigned short* VbT  = (unsigned short*)(ws + 136 * MB);  // [2048][4096] 16MB
  float*          S32  = (float*)         (ws + 152 * MB);  // [4096][4096] 64MB
  unsigned short* Pb   = (unsigned short*)(ws + 216 * MB);  // 32MB
  float*          A32  = (float*)         (ws + 152 * MB);  // reuse S32
  float*          X1   = (float*)         (ws + 88 * MB);   // reuse Qb+Kb
  unsigned short* X1b  = (unsigned short*)(ws + 120 * MB);  // reuse Vb
  unsigned short* H1b  = (unsigned short*)(ws + 72 * MB);   // reuse Xb (8MB)
  float*          H2   = (float*)         (ws + 40 * MB);   // reuse X32
  unsigned short* X2b  = (unsigned short*)(ws + 152 * MB);  // reuse A32
  unsigned short* Ybb  = (unsigned short*)(ws + 168 * MB);  // [4096][1024] 8MB

  const dim3 tb(32, 8);
  // 1) Weight transposes f32 [K][N] -> bf16 [N][K]
  transpose_f32_to_bf16<<<dim3(64, 64, 1), tb, 0, stream>>>(Wq,  WqT,  2048, 2048, 0, 0);
  transpose_f32_to_bf16<<<dim3(64, 64, 1), tb, 0, stream>>>(Wk,  WkT,  2048, 2048, 0, 0);
  transpose_f32_to_bf16<<<dim3(64, 64, 1), tb, 0, stream>>>(Wv,  WvT,  2048, 2048, 0, 0);
  transpose_f32_to_bf16<<<dim3(32, 64, 1), tb, 0, stream>>>(W1,  W1T,  2048, 1024, 0, 0);
  transpose_f32_to_bf16<<<dim3(64, 32, 1), tb, 0, stream>>>(W2,  W2T,  1024, 2048, 0, 0);
  transpose_f32_to_bf16<<<dim3(32, 64, 1), tb, 0, stream>>>(Wys, WysT, 2048, 1024, 0, 0);
  transpose_f32_to_bf16<<<dim3(8, 32, 8), tb, 0, stream>>>(Ws, WsT, 1024, 256,
                                                           1024 * 256, 256 * 1024);
  // 2) Pack x
  pack_x_kernel<<<32768, 256, 0, stream>>>(z_stack, X32, Xb);
  // 3) Q, K, V  (M=4096, N=2048, K=2048)
  gemm_bf16_wmma<<<dim3(16, 32, 1), 256, 0, stream>>>(
      Xb, WqT, bq, nullptr, nullptr, Qb, 4096, 2048, 2048, 0, 0, 0, 0);
  gemm_bf16_wmma<<<dim3(16, 32, 1), 256, 0, stream>>>(
      Xb, WkT, bk, nullptr, nullptr, Kb, 4096, 2048, 2048, 0, 0, 0, 0);
  gemm_bf16_wmma<<<dim3(16, 32, 1), 256, 0, stream>>>(
      Xb, WvT, bv, nullptr, nullptr, Vb, 4096, 2048, 2048, 0, 0, 0, 0);
  // 4) V^T for attend
  transpose_bf16<<<dim3(64, 128, 1), tb, 0, stream>>>(Vb, VbT, 4096, 2048);
  // 5) scores = Q @ K^T  (Kb already [N][K]); M=N=4096, K=2048
  gemm_bf16_wmma<<<dim3(32, 32, 1), 256, 0, stream>>>(
      Qb, Kb, nullptr, nullptr, S32, nullptr, 4096, 4096, 2048, 0, 0, 0, 0);
  // 6) softmax rows
  softmax_rows_4096<<<4096, 256, 0, stream>>>(S32, Pb);
  // 7) attended = P @ V  (Bt = V^T [2048][4096]); M=4096, N=2048, K=4096
  gemm_bf16_wmma<<<dim3(16, 32, 1), 256, 0, stream>>>(
      Pb, VbT, nullptr, nullptr, A32, nullptr, 4096, 2048, 4096, 0, 0, 0, 0);
  // 8) x = LN1(x + attended)
  layernorm_rows_2048<<<4096, 256, 0, stream>>>(X32, A32, ln1_g, ln1_b, X1, X1b);
  // 9) h1 = gelu(x @ W1 + b1)   (M=4096, N=1024, K=2048)
  gemm_bf16_wmma<<<dim3(8, 32, 1), 256, 0, stream>>>(
      X1b, W1T, b1, nullptr, nullptr, H1b, 4096, 1024, 2048, 1, 0, 0, 0);
  // 10) h2 = x + h1 @ W2 + b2   (M=4096, N=2048, K=1024)
  gemm_bf16_wmma<<<dim3(16, 32, 1), 256, 0, stream>>>(
      H1b, W2T, b2, X1, H2, nullptr, 4096, 2048, 1024, 0, 0, 0, 0);
  // 11) h = LN2(h2)
  layernorm_rows_2048<<<4096, 256, 0, stream>>>(H2, nullptr, ln2_g, ln2_b,
                                                nullptr, X2b);
  // 12) yb = h @ Wys + bys      (M=4096, N=1024, K=2048)
  gemm_bf16_wmma<<<dim3(8, 32, 1), 256, 0, stream>>>(
      X2b, WysT, bys, nullptr, nullptr, Ybb, 4096, 1024, 2048, 0, 0, 0, 0);
  // 13) y[m] = yb @ Ws[m] + bs[m]  (batched: M=4096, N=256, K=1024, 8 modules)
  gemm_bf16_wmma<<<dim3(2, 32, 8), 256, 0, stream>>>(
      Ybb, WsT, bs, nullptr, out, nullptr, 4096, 256, 1024, 0,
      (long long)256 * 1024, 256, (long long)4096 * 256);
}
